// NSA_GPT_73375221285370
// MI455X (gfx1250) — matrix operand order repaired and checked by reading, not compiled, not run on hardware
//
#include <hip/hip_runtime.h>
#include <hip/hip_bf16.h>
#include <math.h>

// ---------------- model constants ----------------
constexpr int S_   = 1024;
constexpr int D_   = 768;
constexpr int H_   = 6;
constexpr int DH_  = 128;
constexpr int NB_  = 256;   // compressed blocks (S/CBS)
constexpr int CKP_ = 288;   // padded compressed seq len (1 mem + 256 blocks, mult of 32)
constexpr int VP_  = 50304;
constexpr int NW_  = 32;    // sliding windows (S/32)
constexpr float QSCALE = 0.08838834764831845f; // DH^-0.5

// ---------------- WMMA plumbing ----------------
typedef __attribute__((ext_vector_type(16))) __bf16 v16bf;
typedef __attribute__((ext_vector_type(8)))  float  v8f;

static __device__ __forceinline__ unsigned short f2b(float f) {
  union { float f; unsigned u; } v; v.f = f;
  unsigned r = v.u + 0x7FFFu + ((v.u >> 16) & 1u);  // RNE fp32 -> bf16
  return (unsigned short)(r >> 16);
}
static __device__ __forceinline__ unsigned pack2(float a, float b) {
  return (unsigned)f2b(a) | ((unsigned)f2b(b) << 16);
}

// Fragment load: 16 contiguous bf16 per lane (fragment-ordered LDS) -> 2x ds_load_b128
static __device__ __forceinline__ v16bf frag_ld(const unsigned short* base) {
  union { v16bf v; unsigned u[8]; } r;
  const unsigned* p = (const unsigned*)base;
#pragma unroll
  for (int j = 0; j < 8; ++j) r.u[j] = p[j];
  return r.v;
}

static __device__ __forceinline__ v8f wmma_bf16(v16bf a, v16bf b, v8f c) {
  return __builtin_amdgcn_wmma_f32_16x16x32_bf16(false, a, false, b, (short)0, c, false, false);
}

// Inverse of ISA 7.12.2 A-layout: element (row,k) of a 16x32 tile -> (lane, e)
//   lane = (row&15) + 16*h,  h = (k>>3)&1,  e = (k&7) | ((k&16)>>1)
// Inverse of B-layout: element (k,n) of a 32x16 tile -> (lane, e)
//   lane = (n&15) + 16*h,  h = (k>>3)&1,  e = (k&7) | ((k&16)>>1)

// ---------------- generic WMMA GEMM (128x128x32 tile) ----------------
// C[M,N] = A[M,K] @ op(B); transB: B stored [N,K] row-major.
// flags bit0: accumulate into C; bit1: relu^2 epilogue. grid.z batches (sA/sB/sC).
__global__ __launch_bounds__(256) void k_gemm(
    const float* __restrict__ A, const float* __restrict__ B, float* __restrict__ C,
    int M, int N, int K, int lda, int ldb, int ldc,
    long sA, long sB, long sC, int transB, int flags)
{
  __shared__ unsigned short As[8][32][16];  // [mtile][lane][e]  (fragment order)
  __shared__ unsigned short Bs[8][32][16];  // [ntile][lane][e]
  const int tid  = threadIdx.x;
  const int lane = tid & 31;
  const int w    = tid >> 5;
  const int wm   = w & 3;      // 4 waves over M (32 rows each)
  const int wn   = w >> 2;     // 2 waves over N (64 cols each)
  const long bm0 = (long)blockIdx.y * 128;
  const long bn0 = (long)blockIdx.x * 128;
  A += (long)blockIdx.z * sA;
  B += (long)blockIdx.z * sB;
  C += (long)blockIdx.z * sC;

  v8f acc[2][4];
  v8f z8 = {0.f, 0.f, 0.f, 0.f, 0.f, 0.f, 0.f, 0.f};
#pragma unroll
  for (int im = 0; im < 2; ++im)
#pragma unroll
    for (int in = 0; in < 4; ++in) acc[im][in] = z8;

  for (int k0 = 0; k0 < K; k0 += 32) {
    if (k0 + 32 < K) {               // prefetch next A tile (near locality)
      long pr = bm0 + (tid >> 1);
      if (pr < M) __builtin_prefetch(&A[pr * (long)lda + k0 + 32], 0, 3);
    }
    // ---- stage A tile 128x32 as fragments; quad of 4 k per thread ----
    for (int i = tid; i < 1024; i += 256) {
      int row = i >> 3, kk = (i & 7) * 4;
      long gr = bm0 + row;
      int gc = k0 + kk;
      float q0, q1, q2, q3;
      if (gr < M && gc + 3 < K) {
        const float4 f = *reinterpret_cast<const float4*>(&A[gr * (long)lda + gc]);
        q0 = f.x; q1 = f.y; q2 = f.z; q3 = f.w;
      } else {
        q0 = (gr < M && gc + 0 < K) ? A[gr * (long)lda + gc + 0] : 0.f;
        q1 = (gr < M && gc + 1 < K) ? A[gr * (long)lda + gc + 1] : 0.f;
        q2 = (gr < M && gc + 2 < K) ? A[gr * (long)lda + gc + 2] : 0.f;
        q3 = (gr < M && gc + 3 < K) ? A[gr * (long)lda + gc + 3] : 0.f;
      }
      int h  = (kk >> 3) & 1;
      int e0 = (kk & 7) | ((kk & 16) >> 1);
      int ln = (row & 15) + (h << 4);
      unsigned* p = (unsigned*)&As[row >> 4][ln][e0];
      p[0] = pack2(q0, q1);
      p[1] = pack2(q2, q3);
    }
    // ---- stage B tile 32x128 as fragments ----
    for (int i = tid; i < 1024; i += 256) {
      int n = i & 127, kk = (i >> 7) * 4;
      long gn = bn0 + n;
      int gk = k0 + kk;
      float q0 = 0.f, q1 = 0.f, q2 = 0.f, q3 = 0.f;
      if (gn < N) {
        if (transB) {
          if (gk + 3 < K) {
            const float4 f = *reinterpret_cast<const float4*>(&B[gn * (long)ldb + gk]);
            q0 = f.x; q1 = f.y; q2 = f.z; q3 = f.w;
          } else {
            if (gk + 0 < K) q0 = B[gn * (long)ldb + gk + 0];
            if (gk + 1 < K) q1 = B[gn * (long)ldb + gk + 1];
            if (gk + 2 < K) q2 = B[gn * (long)ldb + gk + 2];
            if (gk + 3 < K) q3 = B[gn * (long)ldb + gk + 3];
          }
        } else {
          if (gk + 0 < K) q0 = B[(long)(gk + 0) * ldb + gn];
          if (gk + 1 < K) q1 = B[(long)(gk + 1) * ldb + gn];
          if (gk + 2 < K) q2 = B[(long)(gk + 2) * ldb + gn];
          if (gk + 3 < K) q3 = B[(long)(gk + 3) * ldb + gn];
        }
      }
      int ln = (n & 15) + (((kk >> 3) & 1) << 4);
      int e0 = (kk & 7) | ((kk & 16) >> 1);
      unsigned* p = (unsigned*)&Bs[n >> 4][ln][e0];
      p[0] = pack2(q0, q1);
      p[1] = pack2(q2, q3);
    }
    __syncthreads();
    v16bf af[2], bfr[4];
    af[0] = frag_ld(&As[wm * 2 + 0][lane][0]);
    af[1] = frag_ld(&As[wm * 2 + 1][lane][0]);
#pragma unroll
    for (int t = 0; t < 4; ++t) bfr[t] = frag_ld(&Bs[wn * 4 + t][lane][0]);
#pragma unroll
    for (int im = 0; im < 2; ++im)
#pragma unroll
      for (int in = 0; in < 4; ++in)
        acc[im][in] = wmma_bf16(af[im], bfr[in], acc[im][in]);
    __syncthreads();
  }
  const int cn = lane & 15;
  const int rh = (lane >> 4) << 3;
#pragma unroll
  for (int im = 0; im < 2; ++im)
#pragma unroll
    for (int in = 0; in < 4; ++in) {
      long col = bn0 + (wn * 4 + in) * 16 + cn;
      if (col >= N) continue;
#pragma unroll
      for (int r = 0; r < 8; ++r) {
        long row = bm0 + (wm * 2 + im) * 16 + rh + r;
        if (row >= M) continue;
        float v = acc[im][in][r];
        if (flags & 2) { v = fmaxf(v, 0.f); v *= v; }
        if (flags & 1) C[row * (long)ldc + col] += v;
        else           C[row * (long)ldc + col]  = v;
      }
    }
}

// ---------------- fused lm_head GEMM + CE-loss stats (128x64 tile) ----------------
// logits = A @ B^T; l = 30*sigmoid(c/7.5) in (0,30) => fixed shift 30 for streaming LSE.
__global__ __launch_bounds__(256) void k_lmhead_loss(
    const float* __restrict__ A, const float* __restrict__ B,
    const int* __restrict__ target, float* __restrict__ sumexp, float* __restrict__ tgtlog,
    int M, int N, int K)
{
  __shared__ unsigned short As[8][32][16];
  __shared__ unsigned short Bs[4][32][16];
  __shared__ float Cs[128 * 64];
  const int tid  = threadIdx.x;
  const int lane = tid & 31;
  const int w    = tid >> 5;
  const int wm   = w & 3;      // 4 waves over M
  const int wn   = w >> 2;     // 2 waves over N (32 cols each)
  const long bm0 = (long)blockIdx.y * 128;
  const long bn0 = (long)blockIdx.x * 64;

  v8f acc[2][2];
  v8f z8 = {0.f, 0.f, 0.f, 0.f, 0.f, 0.f, 0.f, 0.f};
  acc[0][0] = z8; acc[0][1] = z8; acc[1][0] = z8; acc[1][1] = z8;

  for (int k0 = 0; k0 < K; k0 += 32) {
    for (int i = tid; i < 1024; i += 256) {
      int row = i >> 3, kk = (i & 7) * 4;
      long gr = bm0 + row;
      int gc = k0 + kk;
      float q0 = 0.f, q1 = 0.f, q2 = 0.f, q3 = 0.f;
      if (gr < M && gc + 3 < K) {
        const float4 f = *reinterpret_cast<const float4*>(&A[gr * (long)K + gc]);
        q0 = f.x; q1 = f.y; q2 = f.z; q3 = f.w;
      }
      int h  = (kk >> 3) & 1;
      int e0 = (kk & 7) | ((kk & 16) >> 1);
      int ln = (row & 15) + (h << 4);
      unsigned* p = (unsigned*)&As[row >> 4][ln][e0];
      p[0] = pack2(q0, q1);
      p[1] = pack2(q2, q3);
    }
    for (int i = tid; i < 512; i += 256) {
      int n = i & 63, kk = (i >> 6) * 4;
      long gn = bn0 + n;
      int gk = k0 + kk;
      float q0 = 0.f, q1 = 0.f, q2 = 0.f, q3 = 0.f;
      if (gn < N && gk + 3 < K) {
        const float4 f = *reinterpret_cast<const float4*>(&B[gn * (long)K + gk]);
        q0 = f.x; q1 = f.y; q2 = f.z; q3 = f.w;
      }
      int ln = (n & 15) + (((kk >> 3) & 1) << 4);
      int e0 = (kk & 7) | ((kk & 16) >> 1);
      unsigned* p = (unsigned*)&Bs[n >> 4][ln][e0];
      p[0] = pack2(q0, q1);
      p[1] = pack2(q2, q3);
    }
    __syncthreads();
    v16bf a0 = frag_ld(&As[wm * 2 + 0][lane][0]);
    v16bf a1 = frag_ld(&As[wm * 2 + 1][lane][0]);
    v16bf b0 = frag_ld(&Bs[wn * 2 + 0][lane][0]);
    v16bf b1 = frag_ld(&Bs[wn * 2 + 1][lane][0]);
    acc[0][0] = wmma_bf16(a0, b0, acc[0][0]);
    acc[0][1] = wmma_bf16(a0, b1, acc[0][1]);
    acc[1][0] = wmma_bf16(a1, b0, acc[1][0]);
    acc[1][1] = wmma_bf16(a1, b1, acc[1][1]);
    __syncthreads();
  }
  const int cn = lane & 15;
  const int rh = (lane >> 4) << 3;
#pragma unroll
  for (int im = 0; im < 2; ++im)
#pragma unroll
    for (int in = 0; in < 2; ++in)
#pragma unroll
      for (int r = 0; r < 8; ++r) {
        int lr = (wm * 2 + im) * 16 + rh + r;
        int lc = (wn * 2 + in) * 16 + cn;
        long row = bm0 + lr, col = bn0 + lc;
        float l = 30.f / (1.f + expf(-acc[im][in][r] / 7.5f));
        float e = 0.f;
        if (row < M && col < N) {
          e = expf(l - 30.f);
          if (target[row] == (int)col) tgtlog[row] = l;
        }
        Cs[lr * 64 + lc] = e;
      }
  __syncthreads();
  if (tid < 128) {
    float s = 0.f;
#pragma unroll 4
    for (int j = 0; j < 64; ++j) s += Cs[tid * 64 + j];
    long row = bm0 + tid;
    if (row < M) atomicAdd(&sumexp[row], s);
  }
}

__global__ __launch_bounds__(256) void k_loss_final(
    const float* __restrict__ sumexp, const float* __restrict__ tgt, float* __restrict__ out)
{
  __shared__ float red[256];
  int tid = threadIdx.x;
  float p = 0.f;
  for (int s = tid; s < S_; s += 256) p += 30.f + logf(sumexp[s]) - tgt[s];
  red[tid] = p; __syncthreads();
  for (int s = 128; s > 0; s >>= 1) { if (tid < s) red[tid] += red[tid + s]; __syncthreads(); }
  if (tid == 0) out[0] = red[0] / (float)S_;
}

// ---------------- elementwise / norm kernels ----------------
__global__ __launch_bounds__(256) void k_embed(
    const int* __restrict__ seq, const float* __restrict__ emb,
    float* __restrict__ x0, float* __restrict__ x)
{
  const int s = blockIdx.x, tid = threadIdx.x;
  __shared__ float red[256];
  const float* er = emb + (long)seq[s] * D_;
  float ss = 0.f;
  for (int c = tid; c < D_; c += 256) { float v = er[c]; ss += v * v; }
  red[tid] = ss; __syncthreads();
  for (int t = 128; t > 0; t >>= 1) { if (tid < t) red[tid] += red[tid + t]; __syncthreads(); }
  float r = rsqrtf(red[0] / D_ + 1e-6f);
  for (int c = tid; c < D_; c += 256) {
    float v = er[c] * r;
    x0[(long)s * D_ + c] = v;
    x [(long)s * D_ + c] = v;
  }
}

__global__ __launch_bounds__(256) void k_rmsnorm(
    const float* __restrict__ x, float* __restrict__ y, int ncols)
{
  const int row = blockIdx.x, tid = threadIdx.x;
  __shared__ float red[256];
  const float* xr = x + (long)row * ncols;
  float ss = 0.f;
  for (int c = tid; c < ncols; c += 256) { float v = xr[c]; ss += v * v; }
  red[tid] = ss; __syncthreads();
  for (int t = 128; t > 0; t >>= 1) { if (tid < t) red[tid] += red[tid + t]; __syncthreads(); }
  float r = rsqrtf(red[0] / ncols + 1e-6f);
  float* yr = y + (long)row * ncols;
  for (int c = tid; c < ncols; c += 256) yr[c] = xr[c] * r;
}

__global__ void k_lambda(float* __restrict__ x, const float* __restrict__ x0,
                         const float* __restrict__ lam, int n)
{
  int i = blockIdx.x * 256 + threadIdx.x;
  if (i < n) x[i] = lam[0] * x[i] + lam[1] * x0[i];
}

__global__ void k_skipadd(float* __restrict__ x, const float* __restrict__ sk,
                          const float* __restrict__ sw, int n)
{
  int i = blockIdx.x * 256 + threadIdx.x;
  if (i < n) x[i] += sw[0] * sk[i];
}

__global__ void k_copy(float* __restrict__ dst, const float* __restrict__ src, int n)
{
  int i = blockIdx.x * 256 + threadIdx.x;
  if (i < n) dst[i] = src[i];
}

__global__ void k_zero(float* __restrict__ p, int n)
{
  int i = blockIdx.x * 256 + threadIdx.x;
  if (i < n) p[i] = 0.f;
}

// split qkv into per-head [H][S][DH] buffers; q pre-scaled by DH^-0.5
__global__ void k_split_qkv(const float* __restrict__ qkv, float* __restrict__ qh,
                            float* __restrict__ kh, float* __restrict__ vh)
{
  int i = blockIdx.x * 256 + threadIdx.x;
  if (i >= S_ * D_) return;
  int s = i / D_, c = i % D_;
  int h = c >> 7, d = c & 127;
  const float* base = qkv + (long)s * (3 * D_);
  long o = ((long)h * S_ + s) * DH_ + d;
  qh[o] = base[c] * QSCALE;
  kh[o] = base[D_ + c];
  vh[o] = base[2 * D_ + c];
}

// kb/vb = per-block concat of 4 rows + kpos/vpos -> [H][NB][512]
__global__ void k_kbf(const float* __restrict__ kh, const float* __restrict__ vh,
                      const float* __restrict__ kposi, const float* __restrict__ vposi,
                      float* __restrict__ kbf, float* __restrict__ vbf)
{
  int i = blockIdx.x * 256 + threadIdx.x;
  if (i >= H_ * NB_ * 512) return;
  int h = i / (NB_ * 512);
  int rem = i % (NB_ * 512);
  int blk = rem >> 9;
  int t = rem & 511;
  int c = t >> 7, d = t & 127;
  int s = blk * 4 + c;
  long o = ((long)h * S_ + s) * DH_ + d;
  long po = ((long)h * 4 + c) * DH_ + d;
  kbf[i] = kh[o] + kposi[po];
  vbf[i] = vh[o] + vposi[po];
}

// ck/cv row 0 = mem_kv, rows 257..287 = 0 (softmax/gemm padding)
__global__ void k_ckfix(float* __restrict__ ck, float* __restrict__ cv,
                        const float* __restrict__ memk, const float* __restrict__ memv)
{
  int i = blockIdx.x * 256 + threadIdx.x;
  if (i >= H_ * 32 * DH_) return;
  int h = i / (32 * DH_);
  int rem = i % (32 * DH_);
  int r = rem >> 7, d = rem & 127;
  float* ckh = ck + (long)h * CKP_ * DH_;
  float* cvh = cv + (long)h * CKP_ * DH_;
  if (r == 0) {
    ckh[d] = memk[(long)h * DH_ + d];
    cvh[d] = memv[(long)h * DH_ + d];
  } else {
    ckh[(long)(256 + r) * DH_ + d] = 0.f;
    cvh[(long)(256 + r) * DH_ + d] = 0.f;
  }
}

// masked softmax over 257 compressed scores (in place), zero pads 257..287
__global__ __launch_bounds__(256) void k_csoftmax(float* __restrict__ csim)
{
  const int q = blockIdx.x, h = blockIdx.y, tid = threadIdx.x;
  float* row = csim + ((long)h * S_ + q) * CKP_;
  __shared__ float red[256];
  int j0 = tid, j1 = tid + 256;
  float v0 = row[j0];
  if (j0 >= 1 && q < j0 * 4 - 1) v0 = -1e30f;
  float v1 = -1e30f;
  if (j1 < 257) { v1 = row[j1]; if (q < j1 * 4 - 1) v1 = -1e30f; }
  red[tid] = fmaxf(v0, v1); __syncthreads();
  for (int t = 128; t > 0; t >>= 1) { if (tid < t) red[tid] = fmaxf(red[tid], red[tid + t]); __syncthreads(); }
  float mx = red[0]; __syncthreads();
  float e0 = expf(v0 - mx);
  float e1 = (j1 < 257) ? expf(v1 - mx) : 0.f;
  red[tid] = e0 + e1; __syncthreads();
  for (int t = 128; t > 0; t >>= 1) { if (tid < t) red[tid] += red[tid + t]; __syncthreads(); }
  float inv = 1.f / red[0];
  row[j0] = e0 * inv;
  if (j1 < 257) row[j1] = e1 * inv;
  else if (j1 < CKP_) row[j1] = 0.f;
}

// top-4 block selection from compressed attention
__global__ void k_sel(const float* __restrict__ csim, int* __restrict__ sel)
{
  int gid = blockIdx.x * 256 + threadIdx.x;
  if (gid >= H_ * S_) return;
  int h = gid >> 10, q = gid & 1023;
  const float* ca = csim + ((long)h * S_ + q) * CKP_;
  int own = q >> 2;
  float bv[4] = {-1e30f, -1e30f, -1e30f, -1e30f};
  int bi[4] = {0, 0, 0, 0};
  for (int j = 0; j < NB_; ++j) {
    float v = ca[j + 1];
    if (j * 4 > q) v = -1.0f;
    if (j == own)  v = 1e9f;
    int idx;
    if      (v > bv[0]) idx = 0;
    else if (v > bv[1]) idx = 1;
    else if (v > bv[2]) idx = 2;
    else if (v > bv[3]) idx = 3;
    else continue;
    for (int t = 3; t > idx; --t) { bv[t] = bv[t - 1]; bi[t] = bi[t - 1]; }
    bv[idx] = v; bi[idx] = j;
  }
  sel[(long)gid * 4 + 0] = bi[0];
  sel[(long)gid * 4 + 1] = bi[1];
  sel[(long)gid * 4 + 2] = bi[2];
  sel[(long)gid * 4 + 3] = bi[3];
}

// fine attention over selected blocks: 16 keys per query
__global__ __launch_bounds__(128) void k_fsel(
    const float* __restrict__ qh, const float* __restrict__ kh,
    const float* __restrict__ vh, const int* __restrict__ sel, float* __restrict__ fout)
{
  const int h = blockIdx.x >> 10, q = blockIdx.x & 1023, tid = threadIdx.x;
  __shared__ float part[128];
  __shared__ float wgt[16];
  __shared__ int rows[16];
  const int* sp = sel + ((long)h * S_ + q) * 4;
  const float* qp = qh + ((long)h * S_ + q) * DH_;
  int sidx = tid >> 3, sub = tid & 7;
  int kr = sp[sidx >> 2] * 4 + (sidx & 3);
  const float* kp = kh + ((long)h * S_ + kr) * DH_;
  float p = 0.f;
#pragma unroll
  for (int u = 0; u < 16; ++u) { int d = sub * 16 + u; p += qp[d] * kp[d]; }
  part[tid] = p; __syncthreads();
  if (tid < 16) {
    float sc = 0.f;
#pragma unroll
    for (int u = 0; u < 8; ++u) sc += part[tid * 8 + u];
    int r2 = sp[tid >> 2] * 4 + (tid & 3);
    rows[tid] = r2;
    wgt[tid] = (r2 <= q) ? sc : -1e30f;
  }
  __syncthreads();
  if (tid == 0) {
    float mx = -1e30f;
    for (int t = 0; t < 16; ++t) mx = fmaxf(mx, wgt[t]);
    float se = 0.f;
    for (int t = 0; t < 16; ++t) { float e = expf(wgt[t] - mx); wgt[t] = e; se += e; }
    float inv = 1.f / se;
    for (int t = 0; t < 16; ++t) wgt[t] *= inv;
  }
  __syncthreads();
  float acc = 0.f;
#pragma unroll
  for (int t = 0; t < 16; ++t)
    acc += wgt[t] * vh[((long)h * S_ + rows[t]) * DH_ + tid];
  fout[((long)h * S_ + q) * DH_ + tid] = acc;
}

// masked softmax for sliding-window scores: ssim[z][qi][64], z = h*NW + n
__global__ void k_ssoftmax(float* __restrict__ ssim)
{
  int r = blockIdx.x * 256 + threadIdx.x;
  if (r >= H_ * NW_ * 32) return;
  int z = r >> 5, qi = r & 31;
  int n = z & (NW_ - 1);
  float* row = ssim + (long)r * 64;
  float mx = -1e30f;
  for (int j = 0; j < 64; ++j) {
    int kj = j - 32;
    int diff = qi - kj;
    bool ok = (diff >= 0) && (diff < 32) && (n * 32 + kj >= 0);
    float v = ok ? row[j] : -1e30f;
    row[j] = v;
    mx = fmaxf(mx, v);
  }
  float se = 0.f;
  for (int j = 0; j < 64; ++j) { float e = expf(row[j] - mx); row[j] = e; se += e; }
  float inv = 1.f / se;
  for (int j = 0; j < 64; ++j) row[j] *= inv;
}

// gates: sigmoid(x @ gate_w), N=18
__global__ void k_gate(const float* __restrict__ xn, const float* __restrict__ gw,
                       float* __restrict__ g)
{
  int s = blockIdx.x, j = threadIdx.x;
  if (j >= 18) return;
  const float* xr = xn + (long)s * D_;
  float acc = 0.f;
  for (int d = 0; d < D_; ++d) acc += xr[d] * gw[(long)d * 18 + j];
  g[(long)s * 18 + j] = 1.f / (1.f + expf(-acc));
}

// gated combine of three branches back to [S,D]
__global__ void k_combine(const float* __restrict__ g, const float* __restrict__ co,
                          const float* __restrict__ fo, const float* __restrict__ so,
                          float* __restrict__ att)
{
  int i = blockIdx.x * 256 + threadIdx.x;
  if (i >= S_ * D_) return;
  int s = i / D_, c = i % D_;
  int h = c >> 7, d = c & 127;
  long o = ((long)h * S_ + s) * DH_ + d;
  const float* gr = g + (long)s * 18 + h * 3;
  att[i] = gr[0] * co[o] + gr[1] * fo[o] + gr[2] * so[o];
}

// ---------------- host orchestration ----------------
static inline dim3 grids(long n) { return dim3((unsigned)((n + 255) / 256)); }

extern "C" void kernel_launch(void* const* d_in, const int* in_sizes, int n_in,
                              void* d_out, int out_size, void* d_ws, size_t ws_size,
                              hipStream_t stream)
{
  (void)in_sizes; (void)n_in; (void)out_size; (void)ws_size;
  const int*   input_seq  = (const int*)d_in[0];
  const int*   target_seq = (const int*)d_in[1];
  const float* embed      = (const float*)d_in[3];
  const float* lm_head    = (const float*)d_in[4];
  const float* lambdas    = (const float*)d_in[5];
  const float* skip_w     = (const float*)d_in[6];
  const float* qkv_w      = (const float*)d_in[7];
  const float* kpos       = (const float*)d_in[8];
  const float* vpos       = (const float*)d_in[9];
  const float* kcmp       = (const float*)d_in[10];
  const float* vcmp       = (const float*)d_in[11];
  const float* memkv      = (const float*)d_in[12];
  const float* gatew      = (const float*)d_in[13];
  const float* outw       = (const float*)d_in[14];
  const float* fcw        = (const float*)d_in[15];
  const float* projw      = (const float*)d_in[16];

  float* ws = (float*)d_ws;
  size_t o = 0;
  float* x0   = ws + o; o += (size_t)S_ * D_;
  float* x    = ws + o; o += (size_t)S_ * D_;
  float* xn   = ws + o; o += (size_t)S_ * D_;
  float* qkv  = ws + o; o += (size_t)S_ * 3 * D_;
  float* qh   = ws + o; o += (size_t)H_ * S_ * DH_;
  float* kh   = ws + o; o += (size_t)H_ * S_ * DH_;
  float* vh   = ws + o; o += (size_t)H_ * S_ * DH_;
  float* kbf  = ws + o; o += (size_t)H_ * NB_ * 512;
  float* vbf  = ws + o; o += (size_t)H_ * NB_ * 512;
  float* ck   = ws + o; o += (size_t)H_ * CKP_ * DH_;
  float* cv   = ws + o; o += (size_t)H_ * CKP_ * DH_;
  float* csim = ws + o; o += (size_t)H_ * S_ * CKP_;
  float* cout = ws + o; o += (size_t)H_ * S_ * DH_;
  float* fout = ws + o; o += (size_t)H_ * S_ * DH_;
  float* sout = ws + o; o += (size_t)H_ * S_ * DH_;
  float* ssim = ws + o; o += (size_t)H_ * NW_ * 32 * 64;
  float* gbuf = ws + o; o += (size_t)S_ * 18;
  float* att  = ws + o; o += (size_t)S_ * D_;
  float* h1   = ws + o; o += (size_t)S_ * 4 * D_;
  float* skb  = ws + o; o += (size_t)6 * S_ * D_;
  float* sume = ws + o; o += (size_t)S_;
  float* tgtl = ws + o; o += (size_t)S_;
  int*   sel  = (int*)(ws + o); o += (size_t)H_ * S_ * 4;

  auto gemm = [stream](const float* A, const float* B, float* C, int M, int N, int K,
                       int lda, int ldb, int ldc, long sA, long sB, long sC,
                       int transB, int flags, int batch) {
    dim3 g((unsigned)((N + 127) / 128), (unsigned)((M + 127) / 128), (unsigned)batch);
    k_gemm<<<g, 256, 0, stream>>>(A, B, C, M, N, K, lda, ldb, ldc, sA, sB, sC, transB, flags);
  };

  k_embed<<<S_, 256, 0, stream>>>(input_seq, embed, x0, x);

  for (int i = 0; i < 12; ++i) {
    if (i >= 6) {
      int j = i - 6;
      k_skipadd<<<grids(S_ * D_), 256, 0, stream>>>(x, skb + (size_t)(5 - j) * S_ * D_,
                                                    skip_w + j, S_ * D_);
    }
    k_lambda<<<grids(S_ * D_), 256, 0, stream>>>(x, x0, lambdas + 2 * i, S_ * D_);

    if (i != 7) {
      k_rmsnorm<<<S_, 256, 0, stream>>>(x, xn, D_);
      gemm(xn, qkv_w + (size_t)i * D_ * 3 * D_, qkv,
           S_, 3 * D_, D_, D_, 3 * D_, 3 * D_, 0, 0, 0, 0, 0, 1);
      k_split_qkv<<<grids(S_ * D_), 256, 0, stream>>>(qkv, qh, kh, vh);
      k_kbf<<<grids((long)H_ * NB_ * 512), 256, 0, stream>>>(
          kh, vh, kpos + (size_t)i * H_ * 4 * DH_, vpos + (size_t)i * H_ * 4 * DH_, kbf, vbf);
      gemm(kbf, kcmp + (size_t)i * 512 * DH_, ck + DH_,
           NB_, DH_, 512, 512, DH_, DH_, (long)NB_ * 512, 0, (long)CKP_ * DH_, 0, 0, H_);
      gemm(vbf, vcmp + (size_t)i * 512 * DH_, cv + DH_,
           NB_, DH_, 512, 512, DH_, DH_, (long)NB_ * 512, 0, (long)CKP_ * DH_, 0, 0, H_);
      k_ckfix<<<grids((long)H_ * 32 * DH_), 256, 0, stream>>>(
          ck, cv, memkv + (size_t)i * 2 * H_ * DH_, memkv + (size_t)i * 2 * H_ * DH_ + H_ * DH_);
      gemm(qh, ck, csim, S_, 257, DH_, DH_, DH_, CKP_,
           (long)S_ * DH_, (long)CKP_ * DH_, (long)S_ * CKP_, 1, 0, H_);
      { dim3 g(S_, H_); k_csoftmax<<<g, 256, 0, stream>>>(csim); }
      gemm(csim, cv, cout, S_, DH_, CKP_, CKP_, DH_, DH_,
           (long)S_ * CKP_, (long)CKP_ * DH_, (long)S_ * DH_, 0, 0, H_);
      k_sel<<<grids((long)H_ * S_), 256, 0, stream>>>(csim, sel);
      k_fsel<<<H_ * S_, 128, 0, stream>>>(qh, kh, vh, sel, fout);
      // sliding window via batched WMMA GEMMs: z = h*NW + n, window base = z*32 rows
      // kcat/vcat = rows [z*32-32, z*32+32): shifted base, n==0 garbage is masked
      gemm(qh, kh - 32 * DH_, ssim, 32, 64, DH_, DH_, DH_, 64,
           32L * DH_, 32L * DH_, 32L * 64, 1 /*transB*/, 0, H_ * NW_);
      k_ssoftmax<<<grids((long)H_ * NW_ * 32), 256, 0, stream>>>(ssim);
      gemm(ssim, vh - 32 * DH_, sout, 32, DH_, 64, 64, DH_, DH_,
           32L * 64, 32L * DH_, 32L * DH_, 0, 0, H_ * NW_);
      k_gate<<<S_, 32, 0, stream>>>(xn, gatew + (size_t)i * D_ * 18, gbuf);
      k_combine<<<grids(S_ * D_), 256, 0, stream>>>(gbuf, cout, fout, sout, att);
      gemm(att, outw + (size_t)i * D_ * D_, x,
           S_, D_, D_, D_, D_, D_, 0, 0, 0, 0, 1 /*accum*/, 1);
    }

    k_rmsnorm<<<S_, 256, 0, stream>>>(x, xn, D_);
    gemm(xn, fcw + (size_t)i * D_ * 4 * D_, h1,
         S_, 4 * D_, D_, D_, 4 * D_, 4 * D_, 0, 0, 0, 0, 2 /*relu^2*/, 1);
    gemm(h1, projw + (size_t)i * 4 * D_ * D_, x,
         S_, D_, 4 * D_, 4 * D_, D_, D_, 0, 0, 0, 0, 1 /*accum*/, 1);

    if (i < 6) k_copy<<<grids(S_ * D_), 256, 0, stream>>>(skb + (size_t)i * S_ * D_, x, S_ * D_);
  }

  k_rmsnorm<<<S_, 256, 0, stream>>>(x, xn, D_);
  k_zero<<<grids(2 * S_), 256, 0, stream>>>(sume, 2 * S_);   // sume + tgtl contiguous
  {
    dim3 g((unsigned)((VP_ + 63) / 64), (unsigned)((S_ + 127) / 128));
    k_lmhead_loss<<<g, 256, 0, stream>>>(xn, lm_head, target_seq, sume, tgtl, S_, VP_, D_);
  }
  k_loss_final<<<1, 256, 0, stream>>>(sume, tgtl, (float*)d_out);
}
